// RNNModel_28673201668520
// MI455X (gfx1250) — matrix operand order repaired and checked by reading, not compiled, not run on hardware
//
#include <hip/hip_runtime.h>
#include <hip/hip_bf16.h>
#include <math.h>

// Problem constants (match reference)
#define B_    64
#define S_    2048
#define E_    128
#define H_    256
#define V_    61
#define NROWS (B_ * S_)     // 131072
#define G3H   (3 * H_)      // 768

typedef __attribute__((ext_vector_type(16))) __bf16 v16bf;
typedef __attribute__((ext_vector_type(8)))  float  v8f;

// exact param types for the gfx1250 async-to-LDS builtin (decoded from hipcc
// diagnostics: param0 = as1 "int __vector(4)*", param1 = as3 "int __vector(4)*")
typedef __attribute__((vector_size(16))) int gi4_;
typedef __attribute__((address_space(1))) gi4_* gptr_gi4;
typedef __attribute__((address_space(3))) gi4_* lptr_gi4;

// ---------------------------------------------------------------- helpers
__device__ __forceinline__ float sigmoidf_(float x) {
    return 1.0f / (1.0f + __expf(-x));
}

// ---------------------------------------------------------------- small kernels
__global__ void k_f32_to_bf16(const float* __restrict__ src, __bf16* __restrict__ dst, int n) {
    int i = blockIdx.x * blockDim.x + threadIdx.x;
    if (i < n) dst[i] = (__bf16)src[i];
}

// pad w4 (61x128) into 64x128 bf16 with zero rows
__global__ void k_pad_w4(const float* __restrict__ w4, __bf16* __restrict__ dst) {
    int i = blockIdx.x * blockDim.x + threadIdx.x;
    if (i >= 64 * E_) return;
    int n = i / E_;
    dst[i] = (n < V_) ? (__bf16)w4[i] : (__bf16)0.0f;
}

// embedding gather -> bf16 activations
__global__ void k_embed(const int* __restrict__ x, const float* __restrict__ emb,
                        __bf16* __restrict__ e) {
    int i = blockIdx.x * blockDim.x + threadIdx.x;
    if (i >= NROWS * E_) return;
    int row = i >> 7;       // / E_
    int j   = i & (E_ - 1);
    e[i] = (__bf16)emb[x[row] * E_ + j];
}

// ---------------------------------------------------------------- generic bf16 WMMA GEMM
// C[M,N] = A[M,K] (bf16 row-major) * W[N,K]^T (bf16 row-major) + bias
// Block: 256 threads = 8 waves. Wave w -> 16-row stripe; 4 N-tiles per wave (A reuse).
// Block tile: 128(M) x 64(N). Requires M%128==0, N%64==0, K%32==0.
// W K-chunk (64x32 bf16 = 4KB) is staged in LDS via gfx1250 async-to-LDS when
// available (ASYNCcnt + barrier), giving 8x B-fragment reuse across the block.
__global__ __launch_bounds__(256) void k_gemm_bf16(
    const __bf16* __restrict__ A, const __bf16* __restrict__ W,
    const float* __restrict__ bias, float* __restrict__ C,
    int M, int N, int K, int Nvalid) {
    __shared__ __bf16 wstage[64 * 32];   // one K-chunk of the weight block

    const int tid  = threadIdx.x;
    const int wave = tid >> 5;
    const int lane = tid & 31;
    const int l    = lane & 15;
    const int hi   = lane >> 4;      // lane half select (ISA 16-bit A/B layouts)

    const int m0 = blockIdx.x * 128 + wave * 16;
    const int n0 = blockIdx.y * 64;

    // staging assignment: 256 threads x 16B = 4KB chunk
    const int srow  = tid >> 2;      // 0..63 (block-relative weight row)
    const int spart = tid & 3;       // 16B sub-chunk within the 32-element row

    v8f acc[4];
    acc[0] = (v8f){}; acc[1] = (v8f){}; acc[2] = (v8f){}; acc[3] = (v8f){};

    for (int k0 = 0; k0 < K; k0 += 32) {
        // ---- stage W[n0..n0+64) x [k0..k0+32) into LDS ----
        {
            const __bf16* gsrc = W + (size_t)(n0 + srow) * K + k0 + spart * 8;
            __bf16* ldst = wstage + srow * 32 + spart * 8;
#if __has_builtin(__builtin_amdgcn_global_load_async_to_lds_b128)
            __builtin_amdgcn_global_load_async_to_lds_b128(
                (gptr_gi4)(gi4_*)gsrc, (lptr_gi4)(gi4_*)ldst, 0, 0);
#if __has_builtin(__builtin_amdgcn_s_wait_asynccnt)
            __builtin_amdgcn_s_wait_asynccnt(0);
#else
            asm volatile("s_wait_asynccnt 0x0" ::: "memory");
#endif
#else
            *(int4*)ldst = *(const int4*)gsrc;
#endif
        }
        __syncthreads();

        // ---- A fragment: 16x32 bf16 tile at (m0, k0) ----
        // lane<16: row=m0+l, elems 0..7 -> K k0..k0+7, elems 8..15 -> k0+16..23
        // lane>=16: same rows,  elems 0..7 -> k0+8..15, elems 8..15 -> k0+24..31
        const __bf16* arow = A + (size_t)(m0 + l) * K + k0 + hi * 8;
        __builtin_prefetch(arow + 32, 0, 1);     // global_prefetch of next K block
        v16bf a;
        #pragma unroll
        for (int e = 0; e < 8; ++e) a[e] = arow[e];
        #pragma unroll
        for (int e = 0; e < 8; ++e) a[8 + e] = arow[16 + e];

        #pragma unroll
        for (int nt = 0; nt < 4; ++nt) {
            // B fragment = W^T tile from LDS: col n reads wstage[n][hi*8 + e] (e<8), wstage[n][16 + hi*8 + e-8] (e>=8)
            const __bf16* wrow = wstage + (nt * 16 + l) * 32 + hi * 8;
            v16bf b;
            #pragma unroll
            for (int e = 0; e < 8; ++e) { b[e] = wrow[e]; b[8 + e] = wrow[16 + e]; }
            acc[nt] = __builtin_amdgcn_wmma_f32_16x16x32_bf16(
                false, a, false, b, (short)0, acc[nt], false, false);
        }
        __syncthreads();
    }

    // D layout: lane<16 -> M = e, lane>=16 -> M = 8+e ; col = n0 + nt*16 + (lane&15)
    #pragma unroll
    for (int nt = 0; nt < 4; ++nt) {
        const int col = n0 + nt * 16 + l;
        if (col < Nvalid) {
            const float bv = bias ? bias[col] : 0.0f;
            #pragma unroll
            for (int e = 0; e < 8; ++e) {
                const int row = m0 + hi * 8 + e;
                C[(size_t)row * N + col] = acc[nt][e] + bv;
            }
        }
    }
}

// ---------------------------------------------------------------- GRU scan (sequential over S)
// grid = 2 (dir 0 = forward, dir 1 = backward), block = 1024 threads (32 waves).
// h (64x256 bf16) and gh (64x768 f32) live in dynamic LDS.
// Per step: gh = h @ W_hh^T via WMMA (192 tiles, 6/wave), then gate math, h update,
// and masked scatter of the hidden state into the layer output (bf16).
// W_hh (bf16, 384KB) stays L2-resident; per-step re-read is trivial vs 192MB L2.
__global__ __launch_bounds__(1024) void k_gru_scan(
    const float* __restrict__ gi_f, const float* __restrict__ gi_b,
    const __bf16* __restrict__ whh_f, const __bf16* __restrict__ whh_b,
    const float* __restrict__ bhh_f, const float* __restrict__ bhh_b,
    const int* __restrict__ seq_lens,
    __bf16* __restrict__ out, int ldo) {
    extern __shared__ char smem_raw[];
    __bf16* h_s  = (__bf16*)smem_raw;                                     // 64*256*2  = 32 KB
    float*  gh_s = (float*)(smem_raw + (size_t)B_ * H_ * sizeof(__bf16)); // 64*768*4 = 192 KB
    __shared__ int lens_s[B_];

    const int dir = blockIdx.x;
    const float*  gi  = dir ? gi_b  : gi_f;
    const __bf16* whh = dir ? whh_b : whh_f;
    const float*  bhh = dir ? bhh_b : bhh_f;
    const int colOff  = dir * H_;

    const int tid  = threadIdx.x;
    const int wave = tid >> 5;
    const int lane = tid & 31;
    const int l    = lane & 15;
    const int hi   = lane >> 4;

    for (int i = tid; i < B_ * H_; i += 1024) h_s[i] = (__bf16)0.0f;
    if (tid < B_) lens_s[tid] = seq_lens[tid];
    __syncthreads();

    for (int t = 0; t < S_; ++t) {
        // ---- gh = h @ W_hh^T  (64x768, K=256) ----
        for (int T = wave; T < 4 * 48; T += 32) {     // 192 tiles, exactly 6 per wave
            const int mt = T & 3, nt = T >> 2;
            const int m0 = mt * 16, n0 = nt * 16;
            v8f acc = (v8f){};
            for (int k0 = 0; k0 < H_; k0 += 32) {
                const __bf16* arow = h_s + (m0 + l) * H_ + k0 + hi * 8;
                v16bf a;
                #pragma unroll
                for (int e = 0; e < 8; ++e) a[e] = arow[e];
                #pragma unroll
                for (int e = 0; e < 8; ++e) a[8 + e] = arow[16 + e];
                const __bf16* wrow = whh + (size_t)(n0 + l) * H_ + k0 + hi * 8;
                v16bf b;
                #pragma unroll
                for (int e = 0; e < 8; ++e) { b[e] = wrow[e]; b[8 + e] = wrow[16 + e]; }
                acc = __builtin_amdgcn_wmma_f32_16x16x32_bf16(
                    false, a, false, b, (short)0, acc, false, false);
            }
            #pragma unroll
            for (int e = 0; e < 8; ++e)
                gh_s[(m0 + hi * 8 + e) * G3H + n0 + l] = acc[e];
        }
        __syncthreads();

        // ---- gate math + hidden update (16 elems/thread) ----
        for (int i = tid; i < B_ * H_; i += 1024) {
            const int b = i >> 8;            // / H_
            const int j = i & (H_ - 1);
            const int len = lens_s[b];
            if (t < len) {
                const int pos = dir ? (len - 1 - t) : t;   // involution -> direct scatter
                const float* gir = gi + ((size_t)b * S_ + pos) * G3H;
                const float hr = gh_s[b * G3H + j]          + bhh[j];
                const float hz = gh_s[b * G3H + H_ + j]     + bhh[H_ + j];
                const float hn = gh_s[b * G3H + 2 * H_ + j] + bhh[2 * H_ + j];
                const float r  = sigmoidf_(gir[j] + hr);
                const float z  = sigmoidf_(gir[H_ + j] + hz);
                const float nn = tanhf(gir[2 * H_ + j] + r * hn);
                const float hp = (float)h_s[i];
                const float hc = (1.0f - z) * nn + z * hp;
                h_s[i] = (__bf16)hc;
                out[((size_t)b * S_ + pos) * ldo + colOff + j] = (__bf16)hc;
            }
        }
        __syncthreads();
    }
}

// ---------------------------------------------------------------- batchnorm stats (one block / column)
__global__ __launch_bounds__(256) void k_colstats(const float* __restrict__ X, int nrows, int C,
                                                  float* __restrict__ sum, float* __restrict__ sumsq) {
    __shared__ float s1[256], s2[256];
    const int c = blockIdx.x;
    float a = 0.0f, b = 0.0f;
    for (int r = threadIdx.x; r < nrows; r += 256) {
        float v = X[(size_t)r * C + c];
        a += v; b += v * v;
    }
    s1[threadIdx.x] = a; s2[threadIdx.x] = b;
    __syncthreads();
    for (int s = 128; s > 0; s >>= 1) {
        if (threadIdx.x < s) {
            s1[threadIdx.x] += s1[threadIdx.x + s];
            s2[threadIdx.x] += s2[threadIdx.x + s];
        }
        __syncthreads();
    }
    if (threadIdx.x == 0) { sum[c] = s1[0]; sumsq[c] = s2[0]; }
}

// ---------------------------------------------------------------- BN + leaky relu -> bf16
__global__ void k_bn_leaky(const float* __restrict__ X, const float* __restrict__ sum,
                           const float* __restrict__ sumsq, const float* __restrict__ g,
                           const float* __restrict__ be, int C, __bf16* __restrict__ Y) {
    const size_t i = (size_t)blockIdx.x * blockDim.x + threadIdx.x;
    const size_t total = (size_t)NROWS * (size_t)C;
    if (i >= total) return;
    const int c = (int)(i % (size_t)C);
    const float invN = 1.0f / (float)NROWS;
    const float m   = sum[c] * invN;
    const float var = sumsq[c] * invN - m * m;
    float y = g[c] * (X[i] - m) * rsqrtf(var + 1e-5f) + be[c];
    y = (y >= 0.0f) ? y : 0.01f * y;
    Y[i] = (__bf16)y;
}

// ---------------------------------------------------------------- softmax + mask table
// MASK_TABLE rebuilt from reference _counts: row x in [1,20] has zeros on
// [start[x], start[x]+cnt[x]); everything else (and row 0 / invalid t) is NEG.
__constant__ int kMaskStart[21] = {-1, 0, 4, 10, 12, 14, 16, 18, 20, 24, 26,
                                   29, 35, 36, 37, 39, 43, 49, 53, 57, 58};
__constant__ int kMaskCnt[21]   = { 0, 4, 6,  2,  2,  2,  2,  2,  4,  2,  3,
                                    6, 1, 1,  2,  4,  6,  4,  4,  1,  2};

__global__ void k_softmax_mask(const float* __restrict__ logits, const int* __restrict__ x,
                               const int* __restrict__ lens, float* __restrict__ outp) {
    const int i = blockIdx.x * blockDim.x + threadIdx.x;
    if (i >= NROWS) return;
    const int b = i >> 11;           // / S_
    const int s = i & (S_ - 1);
    const float* lr = logits + (size_t)i * 64;    // padded ld = 64
    float mx = lr[0];
    for (int v = 1; v < V_; ++v) mx = fmaxf(mx, lr[v]);
    float p[V_];
    float ssum = 0.0f;
    for (int v = 0; v < V_; ++v) { p[v] = __expf(lr[v] - mx); ssum += p[v]; }
    const float inv = 1.0f / ssum;
    const bool valid = s < lens[b];
    const int xv = x[i];
    const int st = (valid && xv >= 1) ? kMaskStart[xv] : -1;
    const int en = (st >= 0) ? st + kMaskCnt[xv] : -1;
    float* orow = outp + (size_t)i * V_;
    for (int v = 0; v < V_; ++v) {
        const float mask = (v >= st && v < en) ? 0.0f : -1000000000.0f;
        orow[v] = p[v] * inv + mask;
    }
}

// ---------------------------------------------------------------- host orchestration
extern "C" void kernel_launch(void* const* d_in, const int* in_sizes, int n_in,
                              void* d_out, int out_size, void* d_ws, size_t ws_size,
                              hipStream_t stream) {
    (void)in_sizes; (void)n_in; (void)out_size; (void)ws_size;

    const int*   x     = (const int*)d_in[0];
    const int*   lens  = (const int*)d_in[1];
    const float* emb   = (const float*)d_in[2];
    const float* wih0f = (const float*)d_in[3];
    const float* whh0f = (const float*)d_in[4];
    const float* bih0f = (const float*)d_in[5];
    const float* bhh0f = (const float*)d_in[6];
    const float* wih0b = (const float*)d_in[7];
    const float* whh0b = (const float*)d_in[8];
    const float* bih0b = (const float*)d_in[9];
    const float* bhh0b = (const float*)d_in[10];
    const float* wih1f = (const float*)d_in[11];
    const float* whh1f = (const float*)d_in[12];
    const float* bih1f = (const float*)d_in[13];
    const float* bhh1f = (const float*)d_in[14];
    const float* wih1b = (const float*)d_in[15];
    const float* whh1b = (const float*)d_in[16];
    const float* bih1b = (const float*)d_in[17];
    const float* bhh1b = (const float*)d_in[18];
    const float* w1 = (const float*)d_in[19]; const float* b1  = (const float*)d_in[20];
    const float* g1 = (const float*)d_in[21]; const float* be1 = (const float*)d_in[22];
    const float* w2 = (const float*)d_in[23]; const float* b2  = (const float*)d_in[24];
    const float* g2 = (const float*)d_in[25]; const float* be2 = (const float*)d_in[26];
    const float* w3 = (const float*)d_in[27]; const float* b3  = (const float*)d_in[28];
    const float* g3 = (const float*)d_in[29]; const float* be3 = (const float*)d_in[30];
    const float* w4 = (const float*)d_in[31]; const float* b4  = (const float*)d_in[32];

    char* ws = (char*)d_ws;
    size_t off = 0;
    auto alloc = [&](size_t bytes) -> char* {
        char* p = ws + off;
        off += (bytes + 255) & ~(size_t)255;
        return p;
    };

    __bf16* e_bf    = (__bf16*)alloc((size_t)NROWS * E_ * 2);
    __bf16* wih0f_b = (__bf16*)alloc((size_t)G3H * E_ * 2);
    __bf16* wih0b_b = (__bf16*)alloc((size_t)G3H * E_ * 2);
    __bf16* wih1f_b = (__bf16*)alloc((size_t)G3H * 512 * 2);
    __bf16* wih1b_b = (__bf16*)alloc((size_t)G3H * 512 * 2);
    __bf16* whh0f_b = (__bf16*)alloc((size_t)G3H * H_ * 2);
    __bf16* whh0b_b = (__bf16*)alloc((size_t)G3H * H_ * 2);
    __bf16* whh1f_b = (__bf16*)alloc((size_t)G3H * H_ * 2);
    __bf16* whh1b_b = (__bf16*)alloc((size_t)G3H * H_ * 2);
    __bf16* w1_b    = (__bf16*)alloc((size_t)128 * 512 * 2);
    __bf16* w2_b    = (__bf16*)alloc((size_t)256 * 128 * 2);
    __bf16* w3_b    = (__bf16*)alloc((size_t)128 * 256 * 2);
    __bf16* w4p     = (__bf16*)alloc((size_t)64 * 128 * 2);
    float*  gi_f    = (float*)alloc((size_t)NROWS * G3H * 4);
    float*  gi_b    = (float*)alloc((size_t)NROWS * G3H * 4);
    __bf16* h1      = (__bf16*)alloc((size_t)NROWS * 512 * 2);
    __bf16* h2      = (__bf16*)alloc((size_t)NROWS * 512 * 2);
    float*  o1      = (float*)alloc((size_t)NROWS * 128 * 4);
    __bf16* o1b     = (__bf16*)alloc((size_t)NROWS * 128 * 2);
    float*  o2      = (float*)alloc((size_t)NROWS * 256 * 4);
    __bf16* o2b     = (__bf16*)alloc((size_t)NROWS * 256 * 2);
    float*  o3      = (float*)alloc((size_t)NROWS * 128 * 4);
    __bf16* o3b     = (__bf16*)alloc((size_t)NROWS * 128 * 2);
    float*  logits  = (float*)alloc((size_t)NROWS * 64 * 4);
    float*  ssum    = (float*)alloc(256 * 4);
    float*  ssq     = (float*)alloc(256 * 4);

    auto cvt = [&](const float* s, __bf16* d, int n) {
        k_f32_to_bf16<<<(n + 255) / 256, 256, 0, stream>>>(s, d, n);
    };
    cvt(wih0f, wih0f_b, G3H * E_);
    cvt(wih0b, wih0b_b, G3H * E_);
    cvt(wih1f, wih1f_b, G3H * 512);
    cvt(wih1b, wih1b_b, G3H * 512);
    cvt(whh0f, whh0f_b, G3H * H_);
    cvt(whh0b, whh0b_b, G3H * H_);
    cvt(whh1f, whh1f_b, G3H * H_);
    cvt(whh1b, whh1b_b, G3H * H_);
    cvt(w1, w1_b, 128 * 512);
    cvt(w2, w2_b, 256 * 128);
    cvt(w3, w3_b, 128 * 256);
    k_pad_w4<<<(64 * 128 + 255) / 256, 256, 0, stream>>>(w4, w4p);

    k_embed<<<(NROWS * E_ + 255) / 256, 256, 0, stream>>>(x, emb, e_bf);

    auto gemm = [&](const __bf16* A, const __bf16* W, const float* bias, float* C,
                    int M, int N, int K, int Nvalid) {
        dim3 g(M / 128, N / 64);
        k_gemm_bf16<<<g, 256, 0, stream>>>(A, W, bias, C, M, N, K, Nvalid);
    };

    const size_t scanLds = (size_t)B_ * H_ * 2 + (size_t)B_ * G3H * 4;  // 32KB + 192KB

    // ---- layer 0 ----
    gemm(e_bf, wih0f_b, bih0f, gi_f, NROWS, G3H, E_, G3H);
    gemm(e_bf, wih0b_b, bih0b, gi_b, NROWS, G3H, E_, G3H);
    (void)hipMemsetAsync(h1, 0, (size_t)NROWS * 512 * 2, stream);
    k_gru_scan<<<2, 1024, scanLds, stream>>>(gi_f, gi_b, whh0f_b, whh0b_b,
                                             bhh0f, bhh0b, lens, h1, 512);

    // ---- layer 1 ----
    gemm(h1, wih1f_b, bih1f, gi_f, NROWS, G3H, 512, G3H);
    gemm(h1, wih1b_b, bih1b, gi_b, NROWS, G3H, 512, G3H);
    (void)hipMemsetAsync(h2, 0, (size_t)NROWS * 512 * 2, stream);
    k_gru_scan<<<2, 1024, scanLds, stream>>>(gi_f, gi_b, whh1f_b, whh1b_b,
                                             bhh1f, bhh1b, lens, h2, 512);

    // ---- MLP head with batchnorm + leaky relu ----
    gemm(h2, w1_b, b1, o1, NROWS, 128, 512, 128);
    k_colstats<<<128, 256, 0, stream>>>(o1, NROWS, 128, ssum, ssq);
    k_bn_leaky<<<(NROWS * 128 + 255) / 256, 256, 0, stream>>>(o1, ssum, ssq, g1, be1, 128, o1b);

    gemm(o1b, w2_b, b2, o2, NROWS, 256, 128, 256);
    k_colstats<<<256, 256, 0, stream>>>(o2, NROWS, 256, ssum, ssq);
    k_bn_leaky<<<(NROWS * 256 + 255) / 256, 256, 0, stream>>>(o2, ssum, ssq, g2, be2, 256, o2b);

    gemm(o2b, w3_b, b3, o3, NROWS, 128, 256, 128);
    k_colstats<<<128, 256, 0, stream>>>(o3, NROWS, 128, ssum, ssq);
    k_bn_leaky<<<(NROWS * 128 + 255) / 256, 256, 0, stream>>>(o3, ssum, ssq, g3, be3, 128, o3b);

    gemm(o3b, w4p, b4, logits, NROWS, 64, 128, V_);

    k_softmax_mask<<<(NROWS + 255) / 256, 256, 0, stream>>>(logits, x, lens, (float*)d_out);
}
